// SSIM_79886391705722
// MI455X (gfx1250) — matrix operand order repaired and checked by reading, not compiled, not run on hardware
//
#include <hip/hip_runtime.h>

typedef __attribute__((ext_vector_type(16))) _Float16 v16h;
typedef __attribute__((ext_vector_type(8)))  float    v8f;

#define IMG       512
#define CB        4
#define OUT_DIM   494          // 512 - 2*4 - 11 + 1
#define BLK_R     32           // output rows per block (2 wave-rows of 16)
#define BLK_C     64           // output cols per block (4 wave-cols of 16)
#define IN_ROWS   42           // BLK_R + 10 halo
#define IN_COLS   80           // BLK_C + 10 halo, padded to 80 (all loaded, clamped)

// 1-D Gaussian tap (ws=11, sigma=1.5), normalized: 1/sum = 0.26601171
__device__ __forceinline__ float gauss_tap(int t) {
    float x = (float)(t - 5);
    return 0.26601171f * __expf(-x * x * (1.0f / 4.5f));
}

__global__ void ssim_zero_kernel(float* out) {
    if (threadIdx.x < 32) out[threadIdx.x] = 0.0f;
}

__global__ __launch_bounds__(256)
void ssim_wmma_kernel(const float* __restrict__ raw,
                      const float* __restrict__ dst,
                      float* __restrict__ out) {
    __shared__ _Float16 sR[IN_ROWS * IN_COLS];
    __shared__ _Float16 sD[IN_ROWS * IN_COLS];
    __shared__ _Float16 sH[8 * 32 * 16];     // per-wave horizontal-pass staging
    __shared__ float    sBlk;

    const int tid     = threadIdx.x;
    const int plane   = blockIdx.z;          // 0..95  (batch*3 + channel)
    const int batch   = plane / 3;
    const int rowBase = blockIdx.y * BLK_R;  // output-row origin of block
    const int colBase = blockIdx.x * BLK_C;  // output-col origin of block
    const size_t planeOff = (size_t)plane * IMG * IMG;

    if (tid == 0) sBlk = 0.0f;

    // ---- cooperative halo load: f32 global -> (x*255) -> f16 LDS, clamped ----
    for (int idx = tid; idx < IN_ROWS * IN_COLS; idx += 256) {
        int r  = idx / IN_COLS;
        int c  = idx - r * IN_COLS;
        int gy = rowBase + r + CB; if (gy > IMG - 1) gy = IMG - 1;
        int gx = colBase + c + CB; if (gx > IMG - 1) gx = IMG - 1;
        float rv = raw[planeOff + (size_t)gy * IMG + gx] * 255.0f;
        float dv = dst[planeOff + (size_t)gy * IMG + gx] * 255.0f;
        sR[idx] = (_Float16)rv;
        sD[idx] = (_Float16)dv;
    }
    __syncthreads();

    const int lane = tid & 31;
    const int wave = tid >> 5;
    const int wy   = wave >> 2;              // 0..1
    const int wx   = wave & 3;               // 0..3
    const int ty0  = wy * 16;                // tile origin (block-local)
    const int tx0  = wx * 16;

    const int g = lane >> 4;                 // lane half-group
    const int m = lane & 15;                 // A-row / B-col / D-col index

    // ---- static banded weight fragments ----
    // Bw (B operand, 32x16): Bw[K,n] = w[K-n];  B layout: K = e + 8*g + (e>=8 ? 8:0), n = m
    // Av (A operand, 16x32): Av[y,K] = w[K-y];  A layout: K = e + 8*g + (e>=8 ? 8:0), y = m
    union { _Float16 h[16]; v16h v; } Bw, Av;
#pragma unroll
    for (int e = 0; e < 16; ++e) {
        int kB = e + 8 * g + ((e >= 8) ? 8 : 0);
        int tB = kB - m;
        Bw.h[e] = (tB >= 0 && tB <= 10) ? (_Float16)gauss_tap(tB) : (_Float16)0.0f;
        int kA = e + 8 * g + ((e >= 8) ? 8 : 0);
        int tA = kA - m;
        Av.h[e] = (tA >= 0 && tA <= 10) ? (_Float16)gauss_tap(tA) : (_Float16)0.0f;
    }

    // ---- gather input A-fragments once (two row-halves, shared by 5 signals) ----
    _Float16 rf[2][16], df[2][16];
#pragma unroll
    for (int hp = 0; hp < 2; ++hp) {
        int ir = ty0 + hp * 16 + m;
        if (ir > IN_ROWS - 1) ir = IN_ROWS - 1;     // clamp: keeps data finite
        const _Float16* rRow = &sR[ir * IN_COLS];
        const _Float16* dRow = &sD[ir * IN_COLS];
#pragma unroll
        for (int e = 0; e < 16; ++e) {
            int kA = e + 8 * g + ((e >= 8) ? 8 : 0);
            rf[hp][e] = rRow[tx0 + kA];             // tx0+kA <= 48+31 = 79 < 80
            df[hp][e] = dRow[tx0 + kA];
        }
    }

    _Float16* Hbuf = &sH[wave * (32 * 16)];
    v8f V[5];

    // ---- 5 signals x (2 horizontal WMMA + 1 vertical WMMA) ----
#pragma unroll
    for (int sig = 0; sig < 5; ++sig) {
        v8f hacc[2];
#pragma unroll
        for (int hp = 0; hp < 2; ++hp) {
            union { _Float16 h[16]; v16h v; } a;
#pragma unroll
            for (int e = 0; e < 16; ++e) {
                _Float16 r = rf[hp][e];
                _Float16 d = df[hp][e];
                a.h[e] = (sig == 0) ? r
                       : (sig == 1) ? d
                       : (sig == 2) ? (_Float16)(r * r)
                       : (sig == 3) ? (_Float16)(d * d)
                                    : (_Float16)(r * d);
            }
            v8f cz = {0.f, 0.f, 0.f, 0.f, 0.f, 0.f, 0.f, 0.f};
            hacc[hp] = __builtin_amdgcn_wmma_f32_16x16x32_f16(
                false, a.v, false, Bw.v, (short)0, cz, false, false);
        }
        // stage H (f16) in this wave's private LDS slice; D layout: row=i+8g, col=m
#pragma unroll
        for (int hp = 0; hp < 2; ++hp)
#pragma unroll
            for (int i = 0; i < 8; ++i)
                Hbuf[(hp * 16 + i + 8 * g) * 16 + m] = (_Float16)hacc[hp][i];

        // regather as B operand: b[K=e+8g+(e>=8?8:0), n=m]
        union { _Float16 h[16]; v16h v; } b;
#pragma unroll
        for (int e = 0; e < 16; ++e)
            b.h[e] = Hbuf[(e + 8 * g + ((e >= 8) ? 8 : 0)) * 16 + m];

        v8f cz2 = {0.f, 0.f, 0.f, 0.f, 0.f, 0.f, 0.f, 0.f};
        V[sig] = __builtin_amdgcn_wmma_f32_16x16x32_f16(
            false, Av.v, false, b.v, (short)0, cz2, false, false);
    }

    // ---- SSIM map + masked reduction ----
    const float C1c = 6.5025f;      // (0.01*255)^2
    const float C2c = 58.5225f;     // (0.03*255)^2
    float acc = 0.0f;
#pragma unroll
    for (int i = 0; i < 8; ++i) {
        int oy = rowBase + ty0 + i + 8 * g;
        int ox = colBase + tx0 + m;
        if (oy < OUT_DIM && ox < OUT_DIM) {
            float mr  = V[0][i];
            float md  = V[1][i];
            float mrr = mr * mr;
            float mdd = md * md;
            float mrd = mr * md;
            float sxx = V[2][i] - mrr;
            float syy = V[3][i] - mdd;
            float sxy = V[4][i] - mrd;
            float num = (2.0f * mrd + C1c) * (2.0f * sxy + C2c);
            float den = (mrr + mdd + C1c) * (sxx + syy + C2c);
            acc += num / den;
        }
    }
#pragma unroll
    for (int off = 16; off > 0; off >>= 1)
        acc += __shfl_down(acc, off, 32);
    if (lane == 0) atomicAdd(&sBlk, acc);
    __syncthreads();
    if (tid == 0)
        atomicAdd(&out[batch], sBlk * (1.0f / (3.0f * (float)OUT_DIM * (float)OUT_DIM)));
}

extern "C" void kernel_launch(void* const* d_in, const int* in_sizes, int n_in,
                              void* d_out, int out_size, void* d_ws, size_t ws_size,
                              hipStream_t stream) {
    (void)in_sizes; (void)n_in; (void)d_ws; (void)ws_size; (void)out_size;
    const float* raw = (const float*)d_in[0];
    const float* dst = (const float*)d_in[1];
    float* out = (float*)d_out;

    ssim_zero_kernel<<<1, 32, 0, stream>>>(out);

    dim3 grid((OUT_DIM + BLK_C - 1) / BLK_C,   // 8 col-blocks
              (OUT_DIM + BLK_R - 1) / BLK_R,   // 16 row-blocks
              96);                             // 32 batches * 3 channels
    ssim_wmma_kernel<<<grid, dim3(256), 0, stream>>>(raw, dst, out);
}
